// NeuralKF_29291676959082
// MI455X (gfx1250) — matrix operand order repaired and checked by reading, not compiled, not run on hardware
//
#include <hip/hip_runtime.h>

// ---------------------------------------------------------------------------
// NeuralKF scan for gfx1250 (MI455X): WMMA f16 GEMMs + per-lane Kalman update
// ---------------------------------------------------------------------------

typedef _Float16 half_t;
typedef __attribute__((ext_vector_type(16))) _Float16 v16h;
typedef __attribute__((ext_vector_type(8)))  _Float16 v8h;
typedef __attribute__((ext_vector_type(8)))  float    v8f;

#define NG_C   4
#define HID_C  64
#define DOUT_C 4
#define DT_C   1.0f
#define PFAC_C 1000.0f

// Wave-level LDS fence: ensure prior ds writes are visible to all lanes of
// this wave before dependent ds reads (cross-lane RAW within one wave).
#define LDS_FENCE() asm volatile("s_wait_dscnt 0" ::: "memory")

// ---------------- LDS layout (bytes) ----------------
// Weights transposed to [N][Kpad] as f16 so a B-fragment is contiguous.
constexpr int OFF_WE1T = 0;                       // [48][32] f16
constexpr int OFF_WE2T = OFF_WE1T + 48 * 32 * 2;  // [48][64] f16
constexpr int OFF_WI0T = OFF_WE2T + 48 * 64 * 2;  // [256][64] f16
constexpr int OFF_WH0T = OFF_WI0T + 256 * 64 * 2; // [256][64] f16
constexpr int OFF_WI1T = OFF_WH0T + 256 * 64 * 2; // [16][64] f16
constexpr int OFF_WH1T = OFF_WI1T + 16 * 64 * 2;  // [16][32] f16
constexpr int OFF_BE1  = OFF_WH1T + 16 * 32 * 2;  // [48] f32
constexpr int OFF_BE2  = OFF_BE1 + 48 * 4;        // [48] f32
constexpr int OFF_B0   = OFF_BE2 + 48 * 4;        // [256] f32
constexpr int OFF_B1   = OFF_B0 + 256 * 4;        // [16] f32
constexpr int OFF_QD   = OFF_B1 + 16 * 4;         // [6] f32 (padded to 8)
constexpr int OFF_RD   = OFF_QD + 8 * 4;          // [4] f32
constexpr int OFF_STG  = ((OFF_RD + 4 * 4) + 63) & ~63;

// Per-wave activation staging (f16 [M=16][Kpad], K-contiguous)
constexpr int STG_FEATS = 0;      // [16][32] f16 = 1024 B
constexpr int STG_E1    = 1024;   // [16][64] f16 = 2048 B
constexpr int STG_E2    = 3072;   // [16][64] f16 = 2048 B
constexpr int STG_H0A   = 5120;   // [16][64] f16 = 2048 B (double buffered)
constexpr int STG_H0B   = 7168;
constexpr int STG_H1A   = 9216;   // [16][32] f16 = 1024 B (double buffered)
constexpr int STG_H1B   = 10240;
constexpr int STG_GBUF  = 11264;  // [16][16] f32 = 1024 B (LSTM1 gates)
constexpr int STG_PER_WAVE = 12288;
constexpr int SMEM_BYTES = OFF_STG + 4 * STG_PER_WAVE;  // 128512 B

__device__ __forceinline__ float sigm(float x) { return 1.0f / (1.0f + __expf(-x)); }
__device__ __forceinline__ float tanh_fast(float x) {
  float e = __expf(2.0f * x);
  return (e - 1.0f) / (e + 1.0f);
}

__device__ __forceinline__ v8f wmma_f16(v16h a, v16h b, v8f c) {
  // D(f32 16x16) = A(f16 16x32) * B(f16 32x16) + C
  return __builtin_amdgcn_wmma_f32_16x16x32_f16(false, a, false, b, (short)0, c,
                                                false, false);
}

// A-fragment: act is [16][kpad] f16 row-major. Lane L: row m=L&15, half=L>>4.
// elems 0..15 -> K = ks*32 + half*16 + e
__device__ __forceinline__ v16h load_a_frag(const half_t* act, int kpad, int ks,
                                            int lane) {
  int m = lane & 15, hf = lane >> 4;
  const half_t* p0 = act + m * kpad + ks * 32 + hf * 16;
  v8h lo = *(const v8h*)p0;
  v8h hi = *(const v8h*)(p0 + 8);
  v16h r;
#pragma unroll
  for (int i = 0; i < 8; ++i) { r[i] = lo[i]; r[i + 8] = hi[i]; }
  return r;
}

// B-fragment: wt is [N][kpad] f16 (transposed weight). Lane L: n = nt*16+(L&15),
// elems e -> K = ks*32 + (L>>4)*16 + e  (16 contiguous halves, 32B aligned)
__device__ __forceinline__ v16h load_b_frag(const half_t* wt, int kpad, int ks,
                                            int nt, int lane) {
  int n = nt * 16 + (lane & 15), hf = lane >> 4;
  return *(const v16h*)(wt + n * kpad + ks * 32 + hf * 16);
}

__global__ void __launch_bounds__(128) nkf_scan_kernel(
    const float* __restrict__ z, const float* __restrict__ x0,
    const float* __restrict__ We1, const float* __restrict__ be1,
    const float* __restrict__ We2, const float* __restrict__ be2,
    const float* __restrict__ Wi0, const float* __restrict__ Wh0,
    const float* __restrict__ b0, const float* __restrict__ Wi1,
    const float* __restrict__ Wh1, const float* __restrict__ b1g,
    const float* __restrict__ Qlog, const float* __restrict__ Rlog,
    float* __restrict__ ll_ws, float* __restrict__ x_ws, int B, int T) {
  extern __shared__ char smem[] __attribute__((aligned(128)));
  const int tid = threadIdx.x;
  const int lane = tid & 31;
  const int wave = tid >> 5;
  const int g = blockIdx.y;

  half_t* we1t = (half_t*)(smem + OFF_WE1T);
  half_t* we2t = (half_t*)(smem + OFF_WE2T);
  half_t* wi0t = (half_t*)(smem + OFF_WI0T);
  half_t* wh0t = (half_t*)(smem + OFF_WH0T);
  half_t* wi1t = (half_t*)(smem + OFF_WI1T);
  half_t* wh1t = (half_t*)(smem + OFF_WH1T);
  float* be1l = (float*)(smem + OFF_BE1);
  float* be2l = (float*)(smem + OFF_BE2);
  float* b0l = (float*)(smem + OFF_B0);
  float* b1ll = (float*)(smem + OFF_B1);
  float* qdl = (float*)(smem + OFF_QD);
  float* rdl = (float*)(smem + OFF_RD);

  // ---- One-time: stage weights f32->f16, transposed [N][Kpad] (zero-pad K) ----
  for (int i = tid; i < 48 * 32; i += 128) {
    int n = i >> 5, k = i & 31;
    we1t[i] = (k < 19) ? (half_t)We1[k * 48 + n] : (half_t)0.0f;
  }
  for (int i = tid; i < 48 * 64; i += 128) {
    int n = i >> 6, k = i & 63;
    we2t[i] = (k < 48) ? (half_t)We2[k * 48 + n] : (half_t)0.0f;
  }
  for (int i = tid; i < 256 * 64; i += 128) {
    int n = i >> 6, k = i & 63;
    wi0t[i] = (k < 48) ? (half_t)Wi0[((g * 48 + k) << 8) + n] : (half_t)0.0f;
  }
  for (int i = tid; i < 256 * 64; i += 128) {
    int n = i >> 6, k = i & 63;
    wh0t[i] = (half_t)Wh0[((g * 64 + k) << 8) + n];
  }
  for (int i = tid; i < 16 * 64; i += 128) {
    int n = i >> 6, k = i & 63;
    wi1t[i] = (half_t)Wi1[((g * 64 + k) << 4) + n];
  }
  for (int i = tid; i < 16 * 32; i += 128) {
    int n = i >> 5, k = i & 31;
    wh1t[i] = (k < 4) ? (half_t)Wh1[((g * 4 + k) << 4) + n] : (half_t)0.0f;
  }
  if (tid < 48) be1l[tid] = be1[tid];
  if (tid < 48) be2l[tid] = be2[tid];
  for (int i = tid; i < 256; i += 128) b0l[i] = b0[g * 256 + i];
  if (tid < 16) b1ll[tid] = b1g[g * 16 + tid];
  if (tid < 6) qdl[tid] = __expf(Qlog[g * 6 + tid]);
  if (tid < 4) rdl[tid] = __expf(Rlog[tid]);
  {  // zero all staging (covers h0=h1=0 init and K zero-padding of act tiles)
    unsigned* p = (unsigned*)(smem + OFF_STG);
    for (int i = tid; i < (4 * STG_PER_WAVE) >> 2; i += 128) p[i] = 0u;
  }
  __syncthreads();

  char* stg = smem + OFF_STG + wave * STG_PER_WAVE;
  half_t* featsL = (half_t*)(stg + STG_FEATS);
  half_t* e1L = (half_t*)(stg + STG_E1);
  half_t* e2L = (half_t*)(stg + STG_E2);
  half_t* h0buf0 = (half_t*)(stg + STG_H0A);
  half_t* h0buf1 = (half_t*)(stg + STG_H0B);
  half_t* h1buf0 = (half_t*)(stg + STG_H1A);
  half_t* h1buf1 = (half_t*)(stg + STG_H1B);
  float* gbufL = (float*)(stg + STG_GBUF);

  const int m = lane & 15;   // batch row within tile / matrix row
  const int hf = lane >> 4;  // wave half
  int b = blockIdx.x * 64 + wave * 16 + m;
  if (b >= B) b = B - 1;

  float Qr[6], Rr[4];
#pragma unroll
  for (int i = 0; i < 6; ++i) Qr[i] = qdl[i];
#pragma unroll
  for (int i = 0; i < 4; ++i) Rr[i] = rdl[i];

  // ---- Per-lane Kalman state (duplicated across wave halves) ----
  float xs[6];
#pragma unroll
  for (int i = 0; i < 6; ++i) xs[i] = x0[b * 6 + i];
  float P[36];
#pragma unroll
  for (int i = 0; i < 36; ++i) P[i] = 0.0f;
#pragma unroll
  for (int i = 0; i < 6; ++i) P[i * 6 + i] = PFAC_C;

  // LSTM cell states: c0 as WMMA C/D fragments (4 hidden tiles x 8 rows), c1 per-lane
  float c0st[4][8];
#pragma unroll
  for (int a = 0; a < 4; ++a)
#pragma unroll
    for (int r = 0; r < 8; ++r) c0st[a][r] = 0.0f;
  float c1st[4] = {0.0f, 0.0f, 0.0f, 0.0f};

  const size_t TB = (size_t)T * B;

  for (int t = 0; t < T; ++t) {
    half_t* h0r = (t & 1) ? h0buf1 : h0buf0;  // prev h0 (read)
    half_t* h0w = (t & 1) ? h0buf0 : h0buf1;  // new h0 (write)
    half_t* h1r = (t & 1) ? h1buf1 : h1buf0;
    half_t* h1w = (t & 1) ? h1buf0 : h1buf1;

    const float* zp = z + ((size_t)b * T + t) * 4;
    float zv[4];
#pragma unroll
    for (int k = 0; k < 4; ++k) zv[k] = zp[k];
    if (t + 1 < T) __builtin_prefetch(zp + 4, 0, 1);  // global_prefetch next z_t

    // ---- Features (19) -> LDS f16 [16][32] (cols 19..31 stay zero) ----
    if (lane < 16) {
      float innov[4];
#pragma unroll
      for (int k = 0; k < 4; ++k) innov[k] = zv[k] - xs[k];
      float ninn = sqrtf(innov[0] * innov[0] + innov[1] * innov[1] +
                         innov[2] * innov[2] + innov[3] * innov[3]);
      float npos = sqrtf(xs[0] * xs[0] + xs[1] * xs[1] + xs[2] * xs[2]);
      float nvel = sqrtf(xs[3] * xs[3] + xs[4] * xs[4] + xs[5] * xs[5]);
      half_t* fr = featsL + m * 32;
#pragma unroll
      for (int k = 0; k < 4; ++k) fr[k] = (half_t)zv[k];
#pragma unroll
      for (int i = 0; i < 6; ++i) fr[4 + i] = (half_t)xs[i];
#pragma unroll
      for (int k = 0; k < 4; ++k) fr[10 + k] = (half_t)innov[k];
      fr[14] = (half_t)ninn;
      fr[15] = (half_t)npos;
      fr[16] = (half_t)nvel;
      fr[17] = (half_t)((float)t * 0.01f);
      fr[18] = (half_t)DT_C;
    }
    LDS_FENCE();

    // ---- GEMM1: e1 = relu(feats @ We1 + be1)   [16x32]x[32x48] ----
    {
      v16h a = load_a_frag(featsL, 32, 0, lane);
#pragma unroll
      for (int nt = 0; nt < 3; ++nt) {
        v8f acc = {};
        acc = wmma_f16(a, load_b_frag(we1t, 32, 0, nt, lane), acc);
        float bb = be1l[nt * 16 + m];
#pragma unroll
        for (int r = 0; r < 8; ++r) {
          float v = acc[r] + bb;
          v = v > 0.0f ? v : 0.0f;
          e1L[(r + 8 * hf) * 64 + nt * 16 + m] = (half_t)v;
        }
      }
    }
    LDS_FENCE();

    // ---- GEMM2: e2 = relu(e1 @ We2 + be2)   [16x64]x[64x48] ----
#pragma unroll
    for (int nt = 0; nt < 3; ++nt) {
      v8f acc = {};
#pragma unroll
      for (int ks = 0; ks < 2; ++ks)
        acc = wmma_f16(load_a_frag(e1L, 64, ks, lane),
                       load_b_frag(we2t, 64, ks, nt, lane), acc);
      float bb = be2l[nt * 16 + m];
#pragma unroll
      for (int r = 0; r < 8; ++r) {
        float v = acc[r] + bb;
        v = v > 0.0f ? v : 0.0f;
        e2L[(r + 8 * hf) * 64 + nt * 16 + m] = (half_t)v;
      }
    }
    LDS_FENCE();

    // ---- GEMM3: LSTM0 gates = e2 @ Wi0 + h0 @ Wh0 + b0  (256 outputs) ----
#pragma unroll
    for (int ht = 0; ht < 4; ++ht) {  // hidden-unit tile: units ht*16..ht*16+15
      v8f gi = {}, gf = {}, gg = {}, go = {};
#pragma unroll
      for (int ks = 0; ks < 2; ++ks) {
        v16h ae = load_a_frag(e2L, 64, ks, lane);
        v16h ah = load_a_frag(h0r, 64, ks, lane);
        gi = wmma_f16(ae, load_b_frag(wi0t, 64, ks, ht, lane), gi);
        gi = wmma_f16(ah, load_b_frag(wh0t, 64, ks, ht, lane), gi);
        gf = wmma_f16(ae, load_b_frag(wi0t, 64, ks, ht + 4, lane), gf);
        gf = wmma_f16(ah, load_b_frag(wh0t, 64, ks, ht + 4, lane), gf);
        gg = wmma_f16(ae, load_b_frag(wi0t, 64, ks, ht + 8, lane), gg);
        gg = wmma_f16(ah, load_b_frag(wh0t, 64, ks, ht + 8, lane), gg);
        go = wmma_f16(ae, load_b_frag(wi0t, 64, ks, ht + 12, lane), go);
        go = wmma_f16(ah, load_b_frag(wh0t, 64, ks, ht + 12, lane), go);
      }
      float bi = b0l[ht * 16 + m], bf = b0l[64 + ht * 16 + m];
      float bg = b0l[128 + ht * 16 + m], bo = b0l[192 + ht * 16 + m];
#pragma unroll
      for (int r = 0; r < 8; ++r) {
        float iv = sigm(gi[r] + bi), fv = sigm(gf[r] + bf);
        float gv = tanh_fast(gg[r] + bg), ov = sigm(go[r] + bo);
        float c = fv * c0st[ht][r] + iv * gv;
        c0st[ht][r] = c;
        float h = ov * tanh_fast(c);
        h0w[(r + 8 * hf) * 64 + ht * 16 + m] = (half_t)h;
      }
    }
    LDS_FENCE();

    // ---- GEMM4: LSTM1 gates = h0n @ Wi1 + h1 @ Wh1 + b1  (16 outputs) ----
    {
      v8f g1 = {};
#pragma unroll
      for (int ks = 0; ks < 2; ++ks)
        g1 = wmma_f16(load_a_frag(h0w, 64, ks, lane),
                      load_b_frag(wi1t, 64, ks, 0, lane), g1);
      g1 = wmma_f16(load_a_frag(h1r, 32, 0, lane),
                    load_b_frag(wh1t, 32, 0, 0, lane), g1);
      float bb = b1ll[m];
#pragma unroll
      for (int r = 0; r < 8; ++r)
        gbufL[(r + 8 * hf) * 16 + m] = g1[r] + bb;
    }
    LDS_FENCE();

    // ---- LSTM1 nonlinearity per lane (i,f,g,o at n = j, j+4, j+8, j+12) ----
    float h1v[4];
#pragma unroll
    for (int j = 0; j < 4; ++j) {
      float ii = gbufL[m * 16 + j];
      float ff = gbufL[m * 16 + 4 + j];
      float g2 = gbufL[m * 16 + 8 + j];
      float oo = gbufL[m * 16 + 12 + j];
      float c = sigm(ff) * c1st[j] + sigm(ii) * tanh_fast(g2);
      c1st[j] = c;
      h1v[j] = sigm(oo) * tanh_fast(c);
    }
    if (lane < 16) {
#pragma unroll
      for (int j = 0; j < 4; ++j) h1w[m * 32 + j] = (half_t)h1v[j];
    }
    LDS_FENCE();

    // ---- Kalman filter update (per-lane, duplicated across wave halves) ----
    float logp = h1v[0];
    float a0 = h1v[1], a1 = h1v[2], a2 = h1v[3];
    float xp[6];
    xp[0] = xs[0] + DT_C * xs[3] + 0.5f * DT_C * DT_C * a0;
    xp[1] = xs[1] + DT_C * xs[4] + 0.5f * DT_C * DT_C * a1;
    xp[2] = xs[2] + DT_C * xs[5] + 0.5f * DT_C * DT_C * a2;
    xp[3] = xs[3] + DT_C * a0;
    xp[4] = xs[4] + DT_C * a1;
    xp[5] = xs[5] + DT_C * a2;

    float Pp[36];
#pragma unroll
    for (int i = 0; i < 6; ++i) {
      float fr6[6];
#pragma unroll
      for (int j = 0; j < 6; ++j)
        fr6[j] = (i < 3) ? (P[i * 6 + j] + DT_C * P[(i + 3) * 6 + j]) : P[i * 6 + j];
#pragma unroll
      for (int j = 0; j < 3; ++j) Pp[i * 6 + j] = fr6[j] + DT_C * fr6[j + 3];
#pragma unroll
      for (int j = 3; j < 6; ++j) Pp[i * 6 + j] = fr6[j];
    }
#pragma unroll
    for (int i = 0; i < 6; ++i) Pp[i * 6 + i] += Qr[i];

    float y[4];
#pragma unroll
    for (int k = 0; k < 4; ++k) y[k] = zv[k] - xp[k];

    // S = Pp[0:4,0:4] + diag(R); Gauss-Jordan inverse + logdet
    float S[16], SI[16];
#pragma unroll
    for (int i = 0; i < 4; ++i)
#pragma unroll
      for (int j = 0; j < 4; ++j) {
        S[i * 4 + j] = Pp[i * 6 + j];
        SI[i * 4 + j] = (i == j) ? 1.0f : 0.0f;
      }
#pragma unroll
    for (int k = 0; k < 4; ++k) S[k * 4 + k] += Rr[k];

    float logdet = 0.0f;
#pragma unroll
    for (int p = 0; p < 4; ++p) {
      float piv = S[p * 4 + p];
      logdet += __logf(piv);
      float inv = 1.0f / piv;
#pragma unroll
      for (int j = 0; j < 4; ++j) { S[p * 4 + j] *= inv; SI[p * 4 + j] *= inv; }
#pragma unroll
      for (int r = 0; r < 4; ++r) {
        if (r != p) {
          float f = S[r * 4 + p];
#pragma unroll
          for (int j = 0; j < 4; ++j) {
            S[r * 4 + j] -= f * S[p * 4 + j];
            SI[r * 4 + j] -= f * SI[p * 4 + j];
          }
        }
      }
    }

    float Siy[4];
#pragma unroll
    for (int k = 0; k < 4; ++k) {
      float s = 0.0f;
#pragma unroll
      for (int j = 0; j < 4; ++j) s += SI[k * 4 + j] * y[j];
      Siy[k] = s;
    }
    float K6[24];  // K = Pp(:,0:4) @ SI
#pragma unroll
    for (int i = 0; i < 6; ++i)
#pragma unroll
      for (int k = 0; k < 4; ++k) {
        float s = 0.0f;
#pragma unroll
        for (int mm = 0; mm < 4; ++mm) s += Pp[i * 6 + mm] * SI[mm * 4 + k];
        K6[i * 4 + k] = s;
      }
#pragma unroll
    for (int i = 0; i < 6; ++i) {
      float s = xp[i];
#pragma unroll
      for (int k = 0; k < 4; ++k) s += K6[i * 4 + k] * y[k];
      xs[i] = s;
    }
#pragma unroll
    for (int i = 0; i < 6; ++i)
#pragma unroll
      for (int j = 0; j < 6; ++j) {
        float s = Pp[i * 6 + j];
#pragma unroll
        for (int k = 0; k < 4; ++k) s -= K6[i * 4 + k] * Pp[k * 6 + j];
        P[i * 6 + j] = s;
      }
    float quad = y[0] * Siy[0] + y[1] * Siy[1] + y[2] * Siy[2] + y[3] * Siy[3];
    float ll = logp - 0.5f * (quad + logdet);

    if (lane < 16) {
      size_t base = (size_t)g * TB + (size_t)t * B + (size_t)b;
      ll_ws[base] = ll;
      float* xo = x_ws + base * 6;
#pragma unroll
      for (int i = 0; i < 6; ++i) xo[i] = xs[i];
    }
  }
}

// ---- Kernel 2: argmax over groups, gather winning x_new into d_out (T,B,6) ----
__global__ void nkf_select_kernel(const float* __restrict__ ll_ws,
                                  const float* __restrict__ x_ws,
                                  float* __restrict__ out, long long total) {
  long long idx = (long long)blockIdx.x * blockDim.x + threadIdx.x;  // t*B + b
  if (idx >= total) return;
  float best = ll_ws[idx];
  int bg = 0;
#pragma unroll
  for (int g = 1; g < NG_C; ++g) {
    float v = ll_ws[(size_t)g * total + idx];
    if (v > best) { best = v; bg = g; }  // strict > : first max wins (argmax)
  }
  const float* xp = x_ws + ((size_t)bg * total + idx) * 6;
  float* op = out + (size_t)idx * 6;
#pragma unroll
  for (int i = 0; i < 6; ++i) op[i] = xp[i];
}

extern "C" void kernel_launch(void* const* d_in, const int* in_sizes, int n_in,
                              void* d_out, int out_size, void* d_ws,
                              size_t ws_size, hipStream_t stream) {
  const float* z    = (const float*)d_in[0];
  const float* x0   = (const float*)d_in[1];
  const float* We1  = (const float*)d_in[2];
  const float* be1  = (const float*)d_in[3];
  const float* We2  = (const float*)d_in[4];
  const float* be2  = (const float*)d_in[5];
  const float* Wi0  = (const float*)d_in[6];
  const float* Wh0  = (const float*)d_in[7];
  const float* b0   = (const float*)d_in[8];
  const float* Wi1  = (const float*)d_in[9];
  const float* Wh1  = (const float*)d_in[10];
  const float* b1g  = (const float*)d_in[11];
  const float* Qlog = (const float*)d_in[12];
  const float* Rlog = (const float*)d_in[13];

  const int B = in_sizes[1] / 6;            // x0 is (B,6)
  const int T = in_sizes[0] / (B * 4);      // z is (B,T,4)

  // Workspace: ll[NG][T*B] then x_new[NG][T*B][6]
  const size_t TB = (size_t)T * B;
  float* ll_ws = (float*)d_ws;
  float* x_ws = ll_ws + (size_t)NG_C * TB;

  dim3 grid1((B + 63) / 64, NG_C);
  nkf_scan_kernel<<<grid1, 128, SMEM_BYTES, stream>>>(
      z, x0, We1, be1, We2, be2, Wi0, Wh0, b0, Wi1, Wh1, b1g, Qlog, Rlog, ll_ws,
      x_ws, B, T);

  long long total = (long long)TB;
  int blocks = (int)((total + 255) / 256);
  nkf_select_kernel<<<blocks, 256, 0, stream>>>(ll_ws, x_ws, (float*)d_out,
                                                total);
}
